// PAB_5970004542110
// MI455X (gfx1250) — matrix operand order repaired and checked by reading, not compiled, not run on hardware
//
#include <hip/hip_runtime.h>

// ---------------------------------------------------------------------------
// MI455X (gfx1250) wave32 WMMA implementation of the position-attention block.
// All GEMMs run on v_wmma_f32_16x16x32_bf16. Global softmax handled flash-style
// (partial max/exp-sum -> combine -> recompute+scale in the AV pass) so the
// 537 MB logits tensor is never materialized.
// ---------------------------------------------------------------------------

typedef __attribute__((ext_vector_type(16))) __bf16 v16bf;
typedef __attribute__((ext_vector_type(2)))  __bf16 bf2;
typedef __attribute__((ext_vector_type(8)))  float  v8f;

constexpr int kB  = 8;
constexpr int kC  = 256;
constexpr int kP  = 64;
constexpr int kHW = 4096;   // 64 * 64

// ---- helpers ---------------------------------------------------------------

__device__ inline __bf16 f2bf(float f) {
  union { float f; unsigned u; } v; v.f = f;
  unsigned r = v.u + 0x7FFFu + ((v.u >> 16) & 1u);   // round-to-nearest-even
  union { unsigned short s; __bf16 b; } o; o.s = (unsigned short)(r >> 16);
  return o.b;
}

__device__ inline __bf16 bfzero() {
  union { unsigned short s; __bf16 b; } o; o.s = 0; return o.b;
}

__device__ inline v8f vzero8() {
  v8f z;
#pragma unroll
  for (int i = 0; i < 8; ++i) z[i] = 0.0f;
  return z;
}

__device__ inline v8f bf16_wmma(v16bf a, v16bf b, v8f c) {
  return __builtin_amdgcn_wmma_f32_16x16x32_bf16(
      /*neg_a=*/false, a, /*neg_b=*/false, b,
      /*c_mod=*/(short)0, c, /*reuse_a=*/false, /*reuse_b=*/false);
}

// A tile (16 rows x 32 K) from row-major src[M][K], stride ld (elements).
// Contiguous k order (k = 16*h + s), as used here: lanes 0-15 row M=lane, K=0..15; lanes 16-31 K=16..31.
__device__ inline v16bf load_A_rm(const __bf16* __restrict__ src, int ld, int lane) {
  int row   = lane & 15;
  int khalf = (lane >> 4) << 4;               // 0 or 16
  const __bf16* p = src + (size_t)row * ld + khalf;
  v16bf a;
#pragma unroll
  for (int v = 0; v < 8; ++v) {
    int k = 2 * v;                            // 0,2,4,...,14
    bf2 t = *(const bf2*)(p + k);
    a[2 * v] = t[0]; a[2 * v + 1] = t[1];
  }
  return a;
}

// B tile (32 K x 16 cols) where element (k, n) lives at src[n*ld + k]
// ("[N][K]" storage -> contiguous 32-bit loads per lane).
// Contiguous k order (k = 16*h + s), as used here: lane = column; lanes 0-15 K=0..15, lanes 16-31 K=16..31.
__device__ inline v16bf load_B_nk(const __bf16* __restrict__ src, int ld, int lane) {
  int col  = lane & 15;
  int koff = (lane >> 4) << 4;                // 0 or 16
  const __bf16* p = src + (size_t)col * ld + koff;
  v16bf b;
#pragma unroll
  for (int v = 0; v < 8; ++v) {
    bf2 t = *(const bf2*)(p + 2 * v);
    b[2 * v] = t[0]; b[2 * v + 1] = t[1];
  }
  return b;
}

// B tile for shifted 3x3-conv input: column n0+lane shifted by (dy,dx) inside
// the 64x64 image; out-of-bounds columns produce zeros (pad=1 semantics).
__device__ inline v16bf load_B_shift(const __bf16* __restrict__ xTb, int lane,
                                     int n0, int k0, int dy, int dx) {
  int col = n0 + (lane & 15);
  int h = col >> 6, w = col & 63;
  int hh = h + dy, ww = w + dx;
  v16bf b;
  if (hh >= 0 && hh < 64 && ww >= 0 && ww < 64) {
    const __bf16* p = xTb + (size_t)((hh << 6) + ww) * kC + k0 + ((lane >> 4) << 4);
#pragma unroll
    for (int v = 0; v < 8; ++v) {
      bf2 t = *(const bf2*)(p + 2 * v);
      b[2 * v] = t[0]; b[2 * v + 1] = t[1];
    }
  } else {
#pragma unroll
    for (int i = 0; i < 16; ++i) b[i] = bfzero();
  }
  return b;
}

// ---- data prep kernels -----------------------------------------------------

// x [B][C][HW] f32 -> xT [B][HW][C] bf16
__global__ void k_make_xT(const float* __restrict__ x, __bf16* __restrict__ xT) {
  size_t i = (size_t)blockIdx.x * blockDim.x + threadIdx.x;   // = b*HW*C + n*C + c
  size_t per = (size_t)kHW * kC;
  size_t b = i / per, rem = i - b * per;
  size_t n = rem / kC, c = rem - n * kC;
  xT[i] = f2bf(x[b * per + c * kHW + n]);
}

__global__ void k_cvt(const float* __restrict__ src, __bf16* __restrict__ dst, int n) {
  int i = blockIdx.x * blockDim.x + threadIdx.x;
  if (i < n) dst[i] = f2bf(src[i]);
}

// OIHW [C][C][3][3] f32 -> [9][Cout][Cin] bf16
__global__ void k_w3(const float* __restrict__ w, __bf16* __restrict__ w9) {
  int i = blockIdx.x * blockDim.x + threadIdx.x;              // t*C*C + o*C + ii
  int t = i / (kC * kC), rem = i % (kC * kC);
  int o = rem / kC, ii = rem % kC;
  w9[i] = f2bf(w[(size_t)(o * kC + ii) * 9 + t]);
}

// ---- 1x1 projections: topT[n][p], cenT[n][p]  (waves 0-3: top, 4-7: center)
__global__ void k_qkproj(const __bf16* __restrict__ xT,
                         const __bf16* __restrict__ wtopB, const __bf16* __restrict__ wcenB,
                         const float* __restrict__ b_top, const float* __restrict__ b_cen,
                         __bf16* __restrict__ topT, __bf16* __restrict__ cenT) {
  int b = blockIdx.y;
  int lane = threadIdx.x & 31, wv = threadIdx.x >> 5;
  int n0 = blockIdx.x * 16;
  const __bf16* Wm  = (wv < 4) ? wtopB : wcenB;
  const float* bias = (wv < 4) ? b_top : b_cen;
  __bf16* outT = ((wv < 4) ? topT : cenT) + (size_t)b * kHW * kP;
  int mt = wv & 3;
  const __bf16* xB = xT + (size_t)b * kHW * kC;
  v8f acc = vzero8();
  for (int k0 = 0; k0 < kC; k0 += 32) {
    v16bf a  = load_A_rm(Wm + (size_t)mt * 16 * kC + k0, kC, lane);
    v16bf bb = load_B_nk(xB + (size_t)n0 * kC + k0, kC, lane);
    acc = bf16_wmma(a, bb, acc);
  }
  int col = lane & 15, hi = lane >> 4;
#pragma unroll
  for (int r = 0; r < 8; ++r) {
    int prow = mt * 16 + r + 8 * hi;
    outT[(size_t)(n0 + col) * kP + prow] = f2bf(acc[r] + bias[prow]);
  }
}

// ---- 3x3 conv as 9 shifted GEMMs, bf16 output in [C][HW] layout ------------
__global__ void k_conv3x3_bf16(const __bf16* __restrict__ xT, const __bf16* __restrict__ W9,
                               const float* __restrict__ bias, __bf16* __restrict__ out) {
  int b = blockIdx.y;
  int lane = threadIdx.x & 31, wv = threadIdx.x >> 5;
  int n0 = blockIdx.x * 16;
  const __bf16* xB = xT + (size_t)b * kHW * kC;
  v8f acc0 = vzero8(), acc1 = vzero8();
  for (int t = 0; t < 9; ++t) {
    int dy = t / 3 - 1, dx = t % 3 - 1;
    for (int k0 = 0; k0 < kC; k0 += 32) {
      v16bf bb = load_B_shift(xB, lane, n0, k0, dy, dx);
      v16bf a0 = load_A_rm(W9 + (size_t)t * kC * kC + (size_t)(wv * 2 + 0) * 16 * kC + k0, kC, lane);
      v16bf a1 = load_A_rm(W9 + (size_t)t * kC * kC + (size_t)(wv * 2 + 1) * 16 * kC + k0, kC, lane);
      acc0 = bf16_wmma(a0, bb, acc0);
      acc1 = bf16_wmma(a1, bb, acc1);
    }
  }
  __bf16* outB = out + (size_t)b * kC * kHW;
  int col = lane & 15, hi = lane >> 4;
#pragma unroll
  for (int r = 0; r < 8; ++r) {
    int c0 = (wv * 2 + 0) * 16 + r + 8 * hi;
    int c1 = (wv * 2 + 1) * 16 + r + 8 * hi;
    outB[(size_t)c0 * kHW + n0 + col] = f2bf(acc0[r] + bias[c0]);
    outB[(size_t)c1 * kHW + n0 + col] = f2bf(acc1[r] + bias[c1]);
  }
}

// ---- same conv, f32 output (final layer writes d_out directly) -------------
__global__ void k_conv3x3_f32(const __bf16* __restrict__ yT, const __bf16* __restrict__ W9,
                              const float* __restrict__ bias, float* __restrict__ out) {
  int b = blockIdx.y;
  int lane = threadIdx.x & 31, wv = threadIdx.x >> 5;
  int n0 = blockIdx.x * 16;
  const __bf16* yB = yT + (size_t)b * kHW * kC;
  v8f acc0 = vzero8(), acc1 = vzero8();
  for (int t = 0; t < 9; ++t) {
    int dy = t / 3 - 1, dx = t % 3 - 1;
    for (int k0 = 0; k0 < kC; k0 += 32) {
      v16bf bb = load_B_shift(yB, lane, n0, k0, dy, dx);
      v16bf a0 = load_A_rm(W9 + (size_t)t * kC * kC + (size_t)(wv * 2 + 0) * 16 * kC + k0, kC, lane);
      v16bf a1 = load_A_rm(W9 + (size_t)t * kC * kC + (size_t)(wv * 2 + 1) * 16 * kC + k0, kC, lane);
      acc0 = bf16_wmma(a0, bb, acc0);
      acc1 = bf16_wmma(a1, bb, acc1);
    }
  }
  float* outB = out + (size_t)b * kC * kHW;
  int col = lane & 15, hi = lane >> 4;
#pragma unroll
  for (int r = 0; r < 8; ++r) {
    int c0 = (wv * 2 + 0) * 16 + r + 8 * hi;
    int c1 = (wv * 2 + 1) * 16 + r + 8 * hi;
    outB[(size_t)c0 * kHW + n0 + col] = acc0[r] + bias[c0];
    outB[(size_t)c1 * kHW + n0 + col] = acc1[r] + bias[c1];
  }
}

// ---- softmax pass 1: per-wave (max, exp-sum) over its 16 logit rows --------
__global__ void k_pass1(const __bf16* __restrict__ cenT, const __bf16* __restrict__ topT,
                        float* __restrict__ part_m, float* __restrict__ part_z) {
  int b = blockIdx.y;
  int lane = threadIdx.x & 31, wv = threadIdx.x >> 5;
  int ntile = blockIdx.x * 4 + wv;
  const __bf16* cenB = cenT + (size_t)b * kHW * kP;
  const __bf16* topB = topT + (size_t)b * kHW * kP;
  int n0 = ntile * 16;
  v16bf a0 = load_A_rm(cenB + (size_t)n0 * kP,      kP, lane);   // K = p 0..31
  v16bf a1 = load_A_rm(cenB + (size_t)n0 * kP + 32, kP, lane);   // K = p 32..63
  float m = -3.0e38f, z = 0.0f;
  for (int m0 = 0; m0 < kHW; m0 += 16) {
    v16bf b0 = load_B_nk(topB + (size_t)m0 * kP,      kP, lane);
    v16bf b1 = load_B_nk(topB + (size_t)m0 * kP + 32, kP, lane);
    v8f acc = vzero8();
    acc = bf16_wmma(a0, b0, acc);
    acc = bf16_wmma(a1, b1, acc);
    float tm = acc[0];
#pragma unroll
    for (int i = 1; i < 8; ++i) tm = fmaxf(tm, acc[i]);
    float nm = fmaxf(m, tm);
    float zs = 0.0f;
#pragma unroll
    for (int i = 0; i < 8; ++i) zs += __expf(acc[i] - nm);
    z = z * __expf(m - nm) + zs;
    m = nm;
  }
  __shared__ float sm[4][32], sz[4][32];
  sm[wv][lane] = m; sz[wv][lane] = z;
  __syncthreads();
  if (lane == 0) {
    float M = sm[wv][0];
#pragma unroll
    for (int i = 1; i < 32; ++i) M = fmaxf(M, sm[wv][i]);
    float Z = 0.0f;
#pragma unroll
    for (int i = 0; i < 32; ++i) Z += sz[wv][i] * __expf(sm[wv][i] - M);
    part_m[b * 256 + ntile] = M;
    part_z[b * 256 + ntile] = Z;
  }
}

// ---- combine partials -> per-batch global (M, Z) ---------------------------
__global__ void k_combine(const float* __restrict__ part_m, const float* __restrict__ part_z,
                          float* __restrict__ Mz) {
  int b = blockIdx.x, t = threadIdx.x;
  __shared__ float sm[256], sz[256];
  sm[t] = part_m[b * 256 + t];
  __syncthreads();
  for (int s = 128; s > 0; s >>= 1) {
    if (t < s) sm[t] = fmaxf(sm[t], sm[t + s]);
    __syncthreads();
  }
  float M = sm[0];
  __syncthreads();
  sz[t] = part_z[b * 256 + t] * __expf(part_m[b * 256 + t] - M);
  __syncthreads();
  for (int s = 128; s > 0; s >>= 1) {
    if (t < s) sz[t] += sz[t + s];
    __syncthreads();
  }
  if (t == 0) { Mz[b * 2] = M; Mz[b * 2 + 1] = sz[0]; }
}

// ---- AV pass: recompute QK tile, exp, LDS C->A relayout, AV WMMA -----------
__global__ void k_attn(const __bf16* __restrict__ cenT, const __bf16* __restrict__ topT,
                       const __bf16* __restrict__ bot, const float* __restrict__ Mz,
                       float* __restrict__ attn) {
  int b = blockIdx.y;
  int lane = threadIdx.x & 31, wv = threadIdx.x >> 5;
  int ntile = blockIdx.x * 4 + (wv >> 1);      // 4 row-tiles per block
  int chalf = (wv & 1) * 128;                  // each wave owns 128 output cols
  const __bf16* cenB = cenT + (size_t)b * kHW * kP;
  const __bf16* topB = topT + (size_t)b * kHW * kP;
  const __bf16* botB = bot  + (size_t)b * kC  * kHW;
  float M = Mz[b * 2], invZ = 1.0f / Mz[b * 2 + 1];
  int n0 = ntile * 16;
  v16bf a0 = load_A_rm(cenB + (size_t)n0 * kP,      kP, lane);
  v16bf a1 = load_A_rm(cenB + (size_t)n0 * kP + 32, kP, lane);
  v8f accO[8];
#pragma unroll
  for (int i = 0; i < 8; ++i) accO[i] = vzero8();
  __shared__ __align__(16) __bf16 Pt[8][16][32];
  int col = lane & 15, hi = lane >> 4;
  for (int m0 = 0; m0 < kHW; m0 += 32) {
    __builtin_prefetch(topB + (size_t)(m0 + 32) * kP, 0, 1);
    v8f L0 = vzero8(), L1 = vzero8();
    {
      v16bf b0 = load_B_nk(topB + (size_t)m0 * kP,             kP, lane);
      v16bf b1 = load_B_nk(topB + (size_t)m0 * kP + 32,        kP, lane);
      L0 = bf16_wmma(a0, b0, L0); L0 = bf16_wmma(a1, b1, L0);
      v16bf b2 = load_B_nk(topB + (size_t)(m0 + 16) * kP,      kP, lane);
      v16bf b3 = load_B_nk(topB + (size_t)(m0 + 16) * kP + 32, kP, lane);
      L1 = bf16_wmma(a0, b2, L1); L1 = bf16_wmma(a1, b3, L1);
    }
    // C-layout (rows striped over VGPRs) -> A-layout via LDS, with exp applied
#pragma unroll
    for (int r = 0; r < 8; ++r) {
      Pt[wv][r + 8 * hi][col]      = f2bf(__expf(L0[r] - M));
      Pt[wv][r + 8 * hi][16 + col] = f2bf(__expf(L1[r] - M));
    }
    v16bf aP = load_A_rm(&Pt[wv][0][0], 32, lane);
#pragma unroll
    for (int ct = 0; ct < 8; ++ct) {
      v16bf bV = load_B_nk(botB + (size_t)(chalf + ct * 16) * kHW + m0, kHW, lane);
      accO[ct] = bf16_wmma(aP, bV, accO[ct]);
    }
  }
  float* attnB = attn + (size_t)b * kHW * kC;
#pragma unroll
  for (int ct = 0; ct < 8; ++ct)
#pragma unroll
    for (int r = 0; r < 8; ++r)
      attnB[(size_t)(n0 + r + 8 * hi) * kC + chalf + ct * 16 + col] = accO[ct][r] * invZ;
}

// ---- residual: y = x + flat-reinterpret(attn); store bf16 [HW][C] ----------
__global__ void k_residual(const float* __restrict__ x, const float* __restrict__ attn,
                           __bf16* __restrict__ yT) {
  size_t i = (size_t)blockIdx.x * blockDim.x + threadIdx.x;   // flat b*C*HW + c*HW + n
  float y = x[i] + attn[i];    // reshape is a pure flat reinterpretation
  size_t per = (size_t)kC * kHW;
  size_t b = i / per, rem = i - b * per;
  size_t c = rem / kHW, n = rem - c * kHW;
  yT[b * per + n * kC + c] = f2bf(y);
}

// ---------------------------------------------------------------------------

extern "C" void kernel_launch(void* const* d_in, const int* in_sizes, int n_in,
                              void* d_out, int out_size, void* d_ws, size_t ws_size,
                              hipStream_t stream) {
  const float* x     = (const float*)d_in[0];
  const float* w_top = (const float*)d_in[1];
  const float* b_top = (const float*)d_in[2];
  const float* w_cen = (const float*)d_in[3];
  const float* b_cen = (const float*)d_in[4];
  const float* w_bot = (const float*)d_in[5];
  const float* b_bot = (const float*)d_in[6];
  const float* w_out = (const float*)d_in[7];
  const float* b_out = (const float*)d_in[8];
  float* out = (float*)d_out;

  char* p = (char*)d_ws;
  auto carve = [&](size_t bytes) {
    char* r = p; p += (bytes + 255) & ~(size_t)255; return (void*)r;
  };
  __bf16* xT    = (__bf16*)carve((size_t)kB * kHW * kC * 2);   // ~16.8 MB
  __bf16* wtopB = (__bf16*)carve((size_t)kP * kC * 2);
  __bf16* wcenB = (__bf16*)carve((size_t)kP * kC * 2);
  __bf16* wbotB = (__bf16*)carve((size_t)9 * kC * kC * 2);
  __bf16* woutB = (__bf16*)carve((size_t)9 * kC * kC * 2);
  __bf16* topT  = (__bf16*)carve((size_t)kB * kHW * kP * 2);   // [B][HW][P]
  __bf16* cenT  = (__bf16*)carve((size_t)kB * kHW * kP * 2);   // [B][HW][P]
  __bf16* bot   = (__bf16*)carve((size_t)kB * kC * kHW * 2);   // [B][C][HW]
  float*  attn  = (float*) carve((size_t)kB * kHW * kC * 4);   // ~33.5 MB
  __bf16* yT    = (__bf16*)carve((size_t)kB * kHW * kC * 2);
  float* part_m = (float*) carve((size_t)kB * 256 * 4);
  float* part_z = (float*) carve((size_t)kB * 256 * 4);
  float* Mz     = (float*) carve((size_t)kB * 2 * 4);

  // prep: bf16 transposed activations + weight repacks
  k_make_xT<<<dim3((unsigned)((size_t)kB * kHW * kC / 256)), 256, 0, stream>>>(x, xT);
  k_cvt<<<dim3((kP * kC + 255) / 256), 256, 0, stream>>>(w_top, wtopB, kP * kC);
  k_cvt<<<dim3((kP * kC + 255) / 256), 256, 0, stream>>>(w_cen, wcenB, kP * kC);
  k_w3 <<<dim3(9 * kC * kC / 256), 256, 0, stream>>>(w_bot, wbotB);
  k_w3 <<<dim3(9 * kC * kC / 256), 256, 0, stream>>>(w_out, woutB);

  // projections + bottom conv (all WMMA)
  k_qkproj      <<<dim3(kHW / 16, kB), 256, 0, stream>>>(xT, wtopB, wcenB, b_top, b_cen, topT, cenT);
  k_conv3x3_bf16<<<dim3(kHW / 16, kB), 256, 0, stream>>>(xT, wbotB, b_bot, bot);

  // global softmax: partial (max, exp-sum) -> combine -> AV with recompute
  k_pass1  <<<dim3(kHW / 64, kB), 128, 0, stream>>>(cenT, topT, part_m, part_z);
  k_combine<<<dim3(kB), 256, 0, stream>>>(part_m, part_z, Mz);
  k_attn   <<<dim3(kHW / 64, kB), 256, 0, stream>>>(cenT, topT, bot, Mz, attn);

  // residual (flat reinterpret) + final conv
  k_residual   <<<dim3((unsigned)((size_t)kB * kC * kHW / 256)), 256, 0, stream>>>(x, attn, yT);
  k_conv3x3_f32<<<dim3(kHW / 16, kB), 256, 0, stream>>>(yT, woutB, b_out, out);
}
